// nh_Block_g_51213190037563
// MI455X (gfx1250) — matrix operand order repaired and checked by reading, not compiled, not run on hardware
//
#include <hip/hip_runtime.h>
#include <hip/hip_bf16.h>

typedef __attribute__((ext_vector_type(16))) _Float16 v16h;
typedef __attribute__((ext_vector_type(8)))  float    v8f;

#define NB   16          // NH neighbors
#define MD   64
#define FF   256
#define HEADS 4
#define DH   16
#define Tn   4096
#define Bn   8
#define NTOK (Bn*Tn)
#define XU   1024        // MD*NH
#define MT6  64          // M-tile of the final GEMM

__device__ inline float lrelu(float x) { return x > 0.f ? x : 0.2f * x; }

// ============ fragment-order packing maps (wave32 WMMA layouts, ISA 7.12.2) ============
__device__ __host__ inline int a_pack_idx(int m, int k) {
  int kt = k >> 5, kk = k & 31;
  int ln = (m & 15) | (((kk >> 3) & 1) << 4);
  int v  = ((kk >> 4) << 2) | ((kk & 7) >> 1);
  int slot = 2 * v + (kk & 1);
  return (kt * 32 + ln) * 16 + slot;
}
__device__ inline v16h load_a16p(const _Float16* base, int kt, int lane) {
  return *(const v16h*)(base + ((size_t)(kt * 32 + lane)) * 16);
}
__device__ inline v16h load_b16p(const _Float16* base, size_t tile, int lane) {
  return *(const v16h*)(base + (tile * 32 + (size_t)lane) * 16);
}

// ---- hand-built fragments for the small K=16-padded attention matmuls ----
__device__ inline v16h load_a16_k16(const _Float16* a, int ld, int lane) {
  int m = lane & 15, hl = lane >> 4;
  v16h r;
#pragma unroll
  for (int v = 0; v < 8; ++v) {
    int kb = 2 * v + hl * 16;
    r[2 * v]     = (kb     < 16) ? a[m * ld + kb]     : (_Float16)0.f;
    r[2 * v + 1] = (kb + 1 < 16) ? a[m * ld + kb + 1] : (_Float16)0.f;
  }
  return r;
}
__device__ inline v16h load_b16_k16(const _Float16* b, int ld, int lane) {
  int n = lane & 15, hl = lane >> 4;
  v16h r;
#pragma unroll
  for (int v = 0; v < 8; ++v) {
    int k = 2 * v + hl * 16;
    r[2 * v]     = (k     < 16) ? b[(k    ) * ld + n] : (_Float16)0.f;
    r[2 * v + 1] = (k + 1 < 16) ? b[(k + 1) * ld + n] : (_Float16)0.f;
  }
  return r;
}
__device__ inline v16h load_bT16_k16(const _Float16* src, int ld, int lane) {
  int n = lane & 15, hl = lane >> 4;
  v16h r;
#pragma unroll
  for (int v = 0; v < 8; ++v) {
    int k = 2 * v + hl * 16;
    r[2 * v]     = (k     < 16) ? src[n * ld + k]     : (_Float16)0.f;
    r[2 * v + 1] = (k + 1 < 16) ? src[n * ld + k + 1] : (_Float16)0.f;
  }
  return r;
}
__device__ inline void store_d16(const v8f& d, float* c, int ld, int lane) {
  int n = lane & 15, hl = lane >> 4;
#pragma unroll
  for (int v = 0; v < 8; ++v) c[(v + 8 * hl) * ld + n] = d[v];
}

#define WMMA(Aa, Bb, Cc) __builtin_amdgcn_wmma_f32_16x16x32_f16(false, (Aa), false, (Bb), (short)0, (Cc), false, false)

// 16x64 @ 64x64 projection (4 head-tiles), bias-add, result -> sT32 (f32 LDS)
__device__ __forceinline__ void proj64(const _Float16* sXhP,
                                       const _Float16* __restrict__ Wp,
                                       const float* __restrict__ bvec,
                                       float* sT32, int lane) {
  int nn = lane & 15;
#pragma unroll
  for (int h = 0; h < HEADS; ++h) {
    v8f acc; for (int j = 0; j < 8; ++j) acc[j] = 0.f;
#pragma unroll
    for (int kt = 0; kt < 2; ++kt) {
      v16h A  = load_a16p(sXhP, kt, lane);
      v16h Bf = load_b16p(Wp, kt * HEADS + h, lane);
      acc = WMMA(A, Bf, acc);
    }
    float bbv = bvec[h * 16 + nn];
#pragma unroll
    for (int v = 0; v < 8; ++v) acc[v] += bbv;
    store_d16(acc, sT32 + h * 16, 64, lane);
  }
}

// ---------------- K0: f32 -> f16 + B-fragment packing ----------------
__global__ void packB_kernel(const float* __restrict__ src, _Float16* __restrict__ dst,
                             int K, int N) {
  int i = blockIdx.x * 256 + threadIdx.x;
  if (i >= K * N) return;
  int k = i / N, n = i % N;
  int kt = k >> 5, kk = k & 31, nt = n >> 4, nn = n & 15;
  int ln = nn | (((kk >> 3) & 1) << 4);
  int slot = ((kk >> 4) << 3) | (kk & 7);
  int NT = N >> 4;
  dst[(((size_t)kt * NT + nt) * 32 + ln) * 16 + slot] = (_Float16)src[i];
}

// ---------------- K1: LayerNorm over MD=64 ----------------
__global__ void ln1_kernel(const float* __restrict__ x, const float* __restrict__ g,
                           const float* __restrict__ be, float* __restrict__ xn) {
  int r = blockIdx.x * 256 + threadIdx.x;
  if (r >= NTOK) return;
  const float* xr = x + (size_t)r * MD;
  float m = 0.f;
  for (int c = 0; c < MD; ++c) m += xr[c];
  m *= (1.f / MD);
  float v = 0.f;
  for (int c = 0; c < MD; ++c) { float d = xr[c] - m; v += d * d; }
  v *= (1.f / MD);
  float inv = rsqrtf(v + 1e-5f);
  float* o = xn + (size_t)r * MD;
  for (int c = 0; c < MD; ++c) o[c] = (xr[c] - m) * inv * g[c] + be[c];
}

// ---------------- K2: kNN (top-16 nearest) ----------------
__global__ __launch_bounds__(256) void knn_kernel(const float* __restrict__ c1g,
                                                  const float* __restrict__ c2g,
                                                  int* __restrict__ idx,
                                                  float* __restrict__ c1o,
                                                  float* __restrict__ c2o) {
  __shared__ float s1[Tn], s2[Tn];
  int tid = threadIdx.x;
  for (int i = tid; i < Tn; i += 256) { s1[i] = c1g[i]; s2[i] = c2g[i]; }
  __syncthreads();
  int t = blockIdx.x * 256 + tid;
  float q1 = s1[t], q2 = s2[t];
  float bd[NB]; int bi[NB];
  for (int j = 0; j < NB; ++j) { bd[j] = 3.4e38f; bi[j] = 0; }
  for (int s = 0; s < Tn; ++s) {
    float d1 = s1[s] - q1, d2 = s2[s] - q2;
    float d = d1 * d1 + d2 * d2;
    if (d < bd[NB - 1]) {
      int p = NB - 1;
      while (p > 0 && bd[p - 1] > d) { bd[p] = bd[p - 1]; bi[p] = bi[p - 1]; --p; }
      bd[p] = d; bi[p] = s;
    }
  }
  for (int j = 0; j < NB; ++j) {
    idx[t * NB + j] = bi[j];
    c1o[t * NB + j] = s1[bi[j]] - q1;
    c2o[t * NB + j] = s2[bi[j]] - q2;
  }
}

// ---------------- K3: relative position bias MLP ----------------
__global__ void bias_kernel(const float* __restrict__ c1, const float* __restrict__ c2,
                            const float* __restrict__ Wp1, const float* __restrict__ bp1,
                            const float* __restrict__ Wp2, const float* __restrict__ bp2,
                            float* __restrict__ bias) {
  long i = (long)blockIdx.x * 256 + threadIdx.x;  // over T*16*16
  if (i >= (long)Tn * NB * NB) return;
  int t = (int)(i >> 8);
  int a = (int)((i >> 4) & 15);
  int bc = (int)(i & 15);
  float p0 = c1[t * NB + a] - c1[t * NB + bc];
  float p1 = c2[t * NB + a] - c2[t * NB + bc];
  float o[HEADS];
  for (int h = 0; h < HEADS; ++h) o[h] = bp2[h];
  for (int d = 0; d < 32; ++d) {
    float hv = lrelu(p0 * Wp1[d] + p1 * Wp1[32 + d] + bp1[d]);
    for (int h = 0; h < HEADS; ++h) o[h] += hv * Wp2[d * HEADS + h];
  }
  for (int h = 0; h < HEADS; ++h)
    bias[(((long)t * HEADS + h) * NB + a) * NB + bc] = o[h];
}

// ---------------- K4: fused per-token attention + FF + LN3 (1 wave / token) ----------------
__global__ __launch_bounds__(32) void attn_token_kernel(
    const float* __restrict__ xn, const int* __restrict__ idx, const float* __restrict__ biasT,
    const _Float16* __restrict__ WqP, const float* __restrict__ bq,
    const _Float16* __restrict__ WkP, const float* __restrict__ bk,
    const _Float16* __restrict__ WvP, const float* __restrict__ bv,
    const float* __restrict__ logit_scale,
    const _Float16* __restrict__ WoP, const float* __restrict__ bo,
    const float* __restrict__ g2, const float* __restrict__ be2,
    const _Float16* __restrict__ Wn1P, const float* __restrict__ bn1,
    const float* __restrict__ Wn2, const float* __restrict__ bn2,
    const float* __restrict__ g3, const float* __restrict__ be3,
    _Float16* __restrict__ xuP) {
  int n = blockIdx.x;              // token in [0, B*T)
  int bb_ = n >> 12;               // / 4096
  int t = n & (Tn - 1);
  int lane = threadIdx.x;
  int hl = lane >> 4, nn = lane & 15;

  __shared__ int   sIdx[NB];
  __shared__ float sScale[HEADS];
  __shared__ __align__(32) _Float16 sXhP[16 * 64];   // packed A (2 ktiles)
  __shared__ float    sXf[16 * 64];
  __shared__ float    sT32[16 * 64];
  __shared__ _Float16 sQ[16 * 64];
  __shared__ _Float16 sK[16 * 64];
  __shared__ _Float16 sV[16 * 64];
  __shared__ float    sAtt[HEADS * 16 * 16];
  __shared__ _Float16 sAttH[HEADS * 16 * 16];
  __shared__ float    sAv[16 * 64];
  __shared__ __align__(32) _Float16 sAvP[16 * 64];   // packed A (2 ktiles)
  __shared__ float sRed[32], sRed2[32], sMV[2];

  if (lane < NB) sIdx[lane] = idx[t * NB + lane];
  if (lane < HEADS) sScale[lane] = expf(fminf(logit_scale[lane], 4.6051702f));
  __syncthreads();

  // gather neighborhood rows (16 x 64): f32 residual + packed-A f16 operand
  for (int e = lane; e < 1024; e += 32) {
    int r = e >> 6, c = e & 63;
    float v = xn[((size_t)bb_ * Tn + sIdx[r]) * MD + c];
    sXf[e] = v;
    sXhP[a_pack_idx(r, c)] = (_Float16)v;
  }
  __syncthreads();

  // ---- Q ----
  proj64(sXhP, WqP, bq, sT32, lane);
  __syncthreads();
  for (int g = lane; g < 64; g += 32) {
    int r = g & 15, h = g >> 4;
    float s = 0.f;
    for (int d = 0; d < DH; ++d) { float v = sT32[r * 64 + h * 16 + d]; s += v * v; }
    float inv = 1.f / fmaxf(sqrtf(s), 1e-6f);
    for (int d = 0; d < DH; ++d)
      sQ[r * 64 + h * 16 + d] = (_Float16)(sT32[r * 64 + h * 16 + d] * inv);
  }
  __syncthreads();
  // ---- K ----
  proj64(sXhP, WkP, bk, sT32, lane);
  __syncthreads();
  for (int g = lane; g < 64; g += 32) {
    int r = g & 15, h = g >> 4;
    float s = 0.f;
    for (int d = 0; d < DH; ++d) { float v = sT32[r * 64 + h * 16 + d]; s += v * v; }
    float inv = 1.f / fmaxf(sqrtf(s), 1e-6f);
    for (int d = 0; d < DH; ++d)
      sK[r * 64 + h * 16 + d] = (_Float16)(sT32[r * 64 + h * 16 + d] * inv);
  }
  __syncthreads();
  // ---- V ----
  proj64(sXhP, WvP, bv, sT32, lane);
  __syncthreads();
  for (int e = lane; e < 1024; e += 32) sV[e] = (_Float16)sT32[e];
  __syncthreads();

  // ---- attention scores: per head one WMMA (K padded 16->32), scale + bias ----
  for (int h = 0; h < HEADS; ++h) {
    v8f acc; for (int j = 0; j < 8; ++j) acc[j] = 0.f;
    v16h A  = load_a16_k16(sQ + h * 16, 64, lane);
    v16h Bf = load_bT16_k16(sK + h * 16, 64, lane);   // B[k][n] = kh[n][k]
    acc = WMMA(A, Bf, acc);
    float sc = sScale[h];
#pragma unroll
    for (int v = 0; v < 8; ++v) {
      int r = v + 8 * hl;
      sAtt[h * 256 + r * 16 + nn] =
          acc[v] * sc + biasT[(((long)t * HEADS + h) * NB + r) * NB + nn];
    }
  }
  __syncthreads();

  // ---- softmax per (head,row) ----
  for (int g = lane; g < 64; g += 32) {
    int h = g >> 4, r = g & 15;
    float* row = sAtt + h * 256 + r * 16;
    float mx = row[0];
    for (int i = 1; i < 16; ++i) mx = fmaxf(mx, row[i]);
    float sum = 0.f;
    for (int i = 0; i < 16; ++i) { float e = expf(row[i] - mx); row[i] = e; sum += e; }
    float inv = 1.f / sum;
    for (int i = 0; i < 16; ++i) sAttH[h * 256 + r * 16 + i] = (_Float16)(row[i] * inv);
  }
  __syncthreads();

  // ---- av = att @ v per head ----
  for (int h = 0; h < HEADS; ++h) {
    v8f acc; for (int j = 0; j < 8; ++j) acc[j] = 0.f;
    v16h A  = load_a16_k16(sAttH + h * 256, 16, lane);
    v16h Bf = load_b16_k16(sV + h * 16, 64, lane);
    acc = WMMA(A, Bf, acc);
    store_d16(acc, sAv + h * 16, 64, lane);
  }
  __syncthreads();
  for (int e = lane; e < 1024; e += 32) {
    int r = e >> 6, c = e & 63;
    sAvP[a_pack_idx(r, c)] = (_Float16)sAv[e];
  }
  __syncthreads();

  // ---- output projection + residual ----
  for (int h = 0; h < HEADS; ++h) {
    v8f acc; for (int j = 0; j < 8; ++j) acc[j] = 0.f;
#pragma unroll
    for (int kt = 0; kt < 2; ++kt) {
      v16h A  = load_a16p(sAvP, kt, lane);
      v16h Bf = load_b16p(WoP, kt * HEADS + h, lane);
      acc = WMMA(A, Bf, acc);
    }
    float bbv = bo[h * 16 + nn];
#pragma unroll
    for (int v = 0; v < 8; ++v) {
      int r = v + 8 * hl;
      sXf[r * 64 + h * 16 + nn] += acc[v] + bbv;
    }
  }
  __syncthreads();

  // ---- LN2 per row (over 64) -> sXf (f32) and packed f16 operand ----
  if (lane < 16) {
    int r = lane;
    float m = 0.f;
    for (int c = 0; c < 64; ++c) m += sXf[r * 64 + c];
    m *= (1.f / 64.f);
    float var = 0.f;
    for (int c = 0; c < 64; ++c) { float d = sXf[r * 64 + c] - m; var += d * d; }
    var *= (1.f / 64.f);
    float inv = rsqrtf(var + 1e-5f);
    for (int c = 0; c < 64; ++c) {
      float v = (sXf[r * 64 + c] - m) * inv * g2[c] + be2[c];
      sXf[r * 64 + c] = v;
      sXhP[a_pack_idx(r, c)] = (_Float16)v;
    }
  }
  __syncthreads();

  // ---- FF: H = lrelu(xfn @ Wn1 + bn1); fold Wn2 dot into epilogue (sT32 scratch) ----
  for (int nt = 0; nt < 16; ++nt) {
    v8f acc; for (int j = 0; j < 8; ++j) acc[j] = 0.f;
#pragma unroll
    for (int kt = 0; kt < 2; ++kt) {
      v16h A  = load_a16p(sXhP, kt, lane);
      v16h Bf = load_b16p(Wn1P, kt * 16 + nt, lane);
      acc = WMMA(A, Bf, acc);
    }
    int col = nt * 16 + nn;
    float bbv = bn1[col];
    float w2 = Wn2[col];
#pragma unroll
    for (int v = 0; v < 8; ++v) {
      float z = lrelu(acc[v] + bbv);
      int slot = (v + 8 * hl) * 64 + nn + ((nt & 3) * 16);   // lane-private slot
      sT32[slot] = (nt < 4) ? z * w2 : sT32[slot] + z * w2;
    }
  }
  __syncthreads();
  // reduce 64 partials per row, add bn2, lrelu, broadcast-add over 64 cols
  if (lane < 16) {
    int r = lane;
    float s = bn2[0];
    for (int c = 0; c < 64; ++c) s += sT32[r * 64 + c];
    s = lrelu(s);
    for (int c = 0; c < 64; ++c) sXf[r * 64 + c] += s;
  }
  __syncthreads();

  // ---- LN3 over 1024, write xu packed in A-fragment order per 16-token tile ----
  {
    float s = 0.f, s2 = 0.f;
    for (int e = lane; e < 1024; e += 32) { float v = sXf[e]; s += v; s2 += v * v; }
    sRed[lane] = s; sRed2[lane] = s2;
  }
  __syncthreads();
  if (lane == 0) {
    float s = 0.f, s2 = 0.f;
    for (int i = 0; i < 32; ++i) { s += sRed[i]; s2 += sRed2[i]; }
    float m = s * (1.f / 1024.f);
    float var = s2 * (1.f / 1024.f) - m * m;
    sMV[0] = m; sMV[1] = rsqrtf(var + 1e-5f);
  }
  __syncthreads();
  float mm = sMV[0], iv = sMV[1];
  int blk = n >> 4, mrow = n & 15;
  _Float16* dstTile = xuP + (size_t)blk * (16 * XU);
  for (int e = lane; e < 1024; e += 32) {
    float v = ((sXf[e] - mm) * iv) * g3[e] + be3[e];
    int kt = e >> 5, kk = e & 31;
    int ln = mrow | (((kk >> 3) & 1) << 4);
    int vv = ((kk >> 4) << 2) | ((kk & 7) >> 1);
    int slot = 2 * vv + (kk & 1);
    dstTile[(kt * 32 + ln) * 16 + slot] = (_Float16)v;
  }
}

// ---------------- K6: out = lrelu(lrelu(xu @ Wu1 + bu1) @ Wu2 + bu2), M-tile 64 ----------------
__global__ __launch_bounds__(256) void final_mlp_kernel(
    const _Float16* __restrict__ xuP, const _Float16* __restrict__ Wu1P,
    const float* __restrict__ bu1, const float* __restrict__ Wu2,
    const float* __restrict__ bu2, float* __restrict__ out) {
  int blk = blockIdx.x;            // rows [blk*64, blk*64+64)
  int tid = threadIdx.x;
  int lane = tid & 31, wave = tid >> 5;
  int hl = lane >> 4, nn = lane & 15;
  __shared__ __align__(32) _Float16 sA[MT6 * XU];  // packed A tile, 128 KB
  __shared__ float sRow[MT6];
  if (tid < MT6) sRow[tid] = 0.f;

  // async copy of pre-packed A tile: global -> LDS (pure memcpy, ASYNCcnt path)
  {
    const _Float16* src = xuP + (size_t)blk * (MT6 * XU);
    unsigned ldsbase = (unsigned)(uintptr_t)(&sA[0]);
#pragma unroll
    for (int it = 0; it < 32; ++it) {
      int elt = (tid + it * 256) * 8;            // 8 halves = 16 bytes per issue
      unsigned laddr = ldsbase + (unsigned)(elt * 2);
      const void* g = (const void*)(src + elt);
      asm volatile("global_load_async_to_lds_b128 %0, %1, off"
                   :: "v"(laddr), "v"(g) : "memory");
    }
  }
  asm volatile("s_wait_asynccnt 0x0" ::: "memory");
  __syncthreads();

  v8f acc[4][4];
#pragma unroll
  for (int mt = 0; mt < 4; ++mt)
    for (int a = 0; a < 4; ++a)
      for (int j = 0; j < 8; ++j) acc[mt][a][j] = 0.f;

  int nbase = wave * 4;            // 4 n-tiles (64 cols) per wave
  for (int kt = 0; kt < 32; ++kt) {
    v16h Bf[4];
#pragma unroll
    for (int a = 0; a < 4; ++a)
      Bf[a] = load_b16p(Wu1P, (size_t)kt * 64 + nbase + a, lane);
#pragma unroll
    for (int mt = 0; mt < 4; ++mt) {
      v16h A = *(const v16h*)(sA + (size_t)mt * (16 * XU) + (size_t)(kt * 32 + lane) * 16);
#pragma unroll
      for (int a = 0; a < 4; ++a)
        acc[mt][a] = WMMA(A, Bf[a], acc[mt][a]);
    }
  }

#pragma unroll
  for (int mt = 0; mt < 4; ++mt) {
    float s[8];
#pragma unroll
    for (int v = 0; v < 8; ++v) s[v] = 0.f;
#pragma unroll
    for (int a = 0; a < 4; ++a) {
      int col = (nbase + a) * 16 + nn;
      float bbv = bu1[col];
      float w2 = Wu2[col];
#pragma unroll
      for (int v = 0; v < 8; ++v) {
        float z = lrelu(acc[mt][a][v] + bbv);
        s[v] += z * w2;
      }
    }
#pragma unroll
    for (int v = 0; v < 8; ++v) atomicAdd(&sRow[mt * 16 + v + 8 * hl], s[v]);
  }
  __syncthreads();
  if (tid < MT6) {
    float z = sRow[tid] + bu2[0];
    out[blk * MT6 + tid] = lrelu(z);
  }
}

// ---------------- host ----------------
extern "C" void kernel_launch(void* const* d_in, const int* in_sizes, int n_in,
                              void* d_out, int out_size, void* d_ws, size_t ws_size,
                              hipStream_t stream) {
  const float* x       = (const float*)d_in[0];
  const float* coords1 = (const float*)d_in[1];
  const float* coords2 = (const float*)d_in[2];
  const float* g1      = (const float*)d_in[3];
  const float* be1     = (const float*)d_in[4];
  const float* Wq      = (const float*)d_in[5];
  const float* bq      = (const float*)d_in[6];
  const float* Wk      = (const float*)d_in[7];
  const float* bk      = (const float*)d_in[8];
  const float* Wv      = (const float*)d_in[9];
  const float* bv      = (const float*)d_in[10];
  const float* lscale  = (const float*)d_in[11];
  const float* Wo      = (const float*)d_in[12];
  const float* bo      = (const float*)d_in[13];
  const float* Wp1     = (const float*)d_in[14];
  const float* bp1     = (const float*)d_in[15];
  const float* Wp2     = (const float*)d_in[16];
  const float* bp2     = (const float*)d_in[17];
  const float* g2      = (const float*)d_in[18];
  const float* be2     = (const float*)d_in[19];
  const float* Wn1     = (const float*)d_in[20];
  const float* bn1     = (const float*)d_in[21];
  const float* Wn2     = (const float*)d_in[22];
  const float* bn2     = (const float*)d_in[23];
  const float* g3      = (const float*)d_in[24];
  const float* be3     = (const float*)d_in[25];
  const float* Wu1     = (const float*)d_in[26];
  const float* bu1     = (const float*)d_in[27];
  const float* Wu2     = (const float*)d_in[28];
  const float* bu2     = (const float*)d_in[29];

  char* ws = (char*)d_ws;
  size_t off = 0;
  auto alloc = [&](size_t bytes) { size_t o = off; off = (off + bytes + 255) & ~(size_t)255; return o; };

  float*    xn    = (float*)(ws + alloc((size_t)NTOK * MD * 4));
  int*      idx   = (int*)(ws + alloc((size_t)Tn * NB * 4));
  float*    c1o   = (float*)(ws + alloc((size_t)Tn * NB * 4));
  float*    c2o   = (float*)(ws + alloc((size_t)Tn * NB * 4));
  float*    biasT = (float*)(ws + alloc((size_t)Tn * HEADS * NB * NB * 4));
  _Float16* WqP   = (_Float16*)(ws + alloc(MD * MD * 2));
  _Float16* WkP   = (_Float16*)(ws + alloc(MD * MD * 2));
  _Float16* WvP   = (_Float16*)(ws + alloc(MD * MD * 2));
  _Float16* WoP   = (_Float16*)(ws + alloc(MD * MD * 2));
  _Float16* Wn1P  = (_Float16*)(ws + alloc(MD * FF * 2));
  _Float16* Wu1P  = (_Float16*)(ws + alloc((size_t)XU * XU * 2));
  _Float16* xuP   = (_Float16*)(ws + alloc((size_t)NTOK * XU * 2));

  // K0: weight conversion + fragment packing
  packB_kernel<<<(MD * MD + 255) / 256, 256, 0, stream>>>(Wq, WqP, MD, MD);
  packB_kernel<<<(MD * MD + 255) / 256, 256, 0, stream>>>(Wk, WkP, MD, MD);
  packB_kernel<<<(MD * MD + 255) / 256, 256, 0, stream>>>(Wv, WvP, MD, MD);
  packB_kernel<<<(MD * MD + 255) / 256, 256, 0, stream>>>(Wo, WoP, MD, MD);
  packB_kernel<<<(MD * FF + 255) / 256, 256, 0, stream>>>(Wn1, Wn1P, MD, FF);
  packB_kernel<<<(XU * XU + 255) / 256, 256, 0, stream>>>(Wu1, Wu1P, XU, XU);

  // K1: LN1
  ln1_kernel<<<(NTOK + 255) / 256, 256, 0, stream>>>(x, g1, be1, xn);
  // K2: kNN
  knn_kernel<<<Tn / 256, 256, 0, stream>>>(coords1, coords2, idx, c1o, c2o);
  // K3: bias MLP
  bias_kernel<<<((long)Tn * NB * NB + 255) / 256, 256, 0, stream>>>(c1o, c2o, Wp1, bp1, Wp2, bp2, biasT);
  // K4: fused per-token attention block (1 wave32 per token)
  attn_token_kernel<<<NTOK, 32, 0, stream>>>(
      xn, idx, biasT, WqP, bq, WkP, bk, WvP, bv, lscale, WoP, bo,
      g2, be2, Wn1P, bn1, Wn2, bn2, g3, be3, xuP);
  // K6: final MLP GEMM (WMMA, M-tile 64, async-LDS staging)
  final_mlp_kernel<<<NTOK / MT6, 256, 0, stream>>>(xuP, Wu1P, bu1, Wu2, bu2, (float*)d_out);
}
